// HeterogeneousSpatialGNN_35802847379677
// MI455X (gfx1250) — matrix operand order repaired and checked by reading, not compiled, not run on hardware
//
#include <hip/hip_runtime.h>
#include <cstdint>

typedef __attribute__((ext_vector_type(16))) _Float16 v16h;
typedef __attribute__((ext_vector_type(8)))  float    v8f;

// Problem constants (fixed by the reference)
constexpr int NN = 100000;   // nodes per type
constexpr int T  = 2;        // node types
constexpr int R  = 2;        // edge types
constexpr int EE = 1000000;  // edges per edge type
constexpr int C  = 64;       // hidden
constexpr int H  = 4;        // heads
constexpr int D  = 16;       // head dim
constexpr int L  = 2;        // layers
constexpr float SCALE = 0.25f;   // 1/sqrt(D)

// ---------------------------------------------------------------------------
// Order-preserving float<->uint encoding for atomic max on f32 scores
// ---------------------------------------------------------------------------
__device__ __forceinline__ unsigned encf(float x) {
  unsigned b = __float_as_uint(x);
  return (b & 0x80000000u) ? ~b : (b | 0x80000000u);
}
__device__ __forceinline__ float decf(unsigned k) {
  unsigned b = (k & 0x80000000u) ? (k & 0x7FFFFFFFu) : ~k;
  return __uint_as_float(b);
}

// ---------------------------------------------------------------------------
// WMMA GEMM: Y[rows,64] = act( X[rows,64] @ W[64,64] + bias )  per type (grid.y)
// mode 0: bias only (k/q/v proj)
// mode 1: bias + ReLU (input proj)
// mode 2: bias + skip blend: Y = beta*val + (1-beta)*Hprev, beta = sigmoid(skip[t])
//
// Block: 128 threads = 4 waves; each wave owns one 16-wide N tile of a 16-row
// M tile. K=64 -> two v_wmma_f32_16x16x32_f16 issues per wave.
// A is staged row-major (lane fragment = two contiguous 16B runs -> ds_load_b128),
// W is staged TRANSPOSED (sBt[n][k]) so each B fragment is also two contiguous
// 16B runs per lane -> ds_load_b128 instead of 16x ds_load_u16 + packing movs.
// Row pitch 72 halves = 144B (16B aligned, and 36 banks/row kills conflicts).
// ---------------------------------------------------------------------------
__global__ __launch_bounds__(128)
void gemm64_wmma_kernel(const float* __restrict__ X, const float* __restrict__ Wt,
                        const float* __restrict__ bias, float* __restrict__ Y,
                        const float* __restrict__ Hprev, const float* __restrict__ skipGate,
                        int rows, int mode)
{
  const int ty = blockIdx.y;
  const size_t matOff = (size_t)ty * rows * C;
  const float* Xm = X + matOff;
  const float* Wm = Wt + (size_t)ty * C * C;
  const float* bm = bias + ty * C;
  float* Ym = Y + matOff;

  __shared__ _Float16 sA [16][72];   // sA[m][k]
  __shared__ _Float16 sBt[64][72];   // sBt[n][k]  (transposed W)
  __shared__ float    sBias[64];

  const int tid = threadIdx.x;
  const int rowBase = blockIdx.x * 16;

  // Stage A tile (16x64 f32 -> f16): 256 float4 loads across 128 threads
  {
    const float4* s4 = (const float4*)(Xm + (size_t)rowBase * C);
    for (int i = tid; i < 256; i += 128) {
      float4 f = s4[i];
      int base = i * 4, r = base >> 6, c = base & 63;
      sA[r][c + 0] = (_Float16)f.x; sA[r][c + 1] = (_Float16)f.y;
      sA[r][c + 2] = (_Float16)f.z; sA[r][c + 3] = (_Float16)f.w;
    }
  }
  // Stage W (64x64 f32, row-major [k][n]) transposed into sBt[n][k]
  {
    const float4* w4 = (const float4*)Wm;
    for (int i = tid; i < 1024; i += 128) {
      float4 f = w4[i];
      int base = i * 4, k = base >> 6, n = base & 63;
      sBt[n + 0][k] = (_Float16)f.x; sBt[n + 1][k] = (_Float16)f.y;
      sBt[n + 2][k] = (_Float16)f.z; sBt[n + 3][k] = (_Float16)f.w;
    }
  }
  if (tid < 64) sBias[tid] = bm[tid];
  __syncthreads();

  const int lane = tid & 31;
  const int wv   = tid >> 5;
  const int half = lane >> 4;     // half-wave select
  const int mr   = lane & 15;     // A row / B col within tile
  const int col  = wv * 16 + mr;  // global column (0..63)

  v8f acc = {};
#pragma unroll
  for (int ko = 0; ko < 2; ++ko) {
    const int kb = ko * 32;
    v16h a, b;
    // A fragment: lane holds row mr; K = kb + {half*8+j, 16+half*8+j}
#pragma unroll
    for (int j = 0; j < 8; ++j) {
      a[j]     = sA[mr][kb + half * 8 + j];
      a[j + 8] = sA[mr][kb + 16 + half * 8 + j];
    }
    // B fragment: lane holds col; K = kb + {half*8+j, 16+half*8+j}  (contiguous runs in sBt)
#pragma unroll
    for (int j = 0; j < 8; ++j) {
      b[j]     = sBt[col][kb + half * 8 + j];
      b[j + 8] = sBt[col][kb + 16 + half * 8 + j];
    }
    acc = __builtin_amdgcn_wmma_f32_16x16x32_f16(false, a, false, b,
                                                 (short)0, acc, false, false);
  }

  float beta = 0.0f;
  if (mode == 2) beta = 1.0f / (1.0f + expf(-skipGate[ty]));

#pragma unroll
  for (int i = 0; i < 8; ++i) {
    int row = rowBase + half * 8 + i;   // D layout: VGPR i <-> M = half*8 + i
    float val = acc[i] + sBias[col];
    if (mode == 1) val = fmaxf(val, 0.0f);
    size_t off = (size_t)row * C + col;
    if (mode == 2) {
      float hp = Hprev[matOff + off];
      val = beta * val + (1.0f - beta) * hp;
    }
    Ym[off] = val;
  }
}

// ---------------------------------------------------------------------------
// Per-(node,head) 16x16 relation transforms: kr = k @ arel[h], vr = v @ mrel[h]
// ---------------------------------------------------------------------------
__global__ __launch_bounds__(256)
void rel_transform_kernel(const float* __restrict__ kin, const float* __restrict__ vin,
                          const float* __restrict__ arel, const float* __restrict__ mrel,
                          float* __restrict__ kr, float* __restrict__ vr, int total)
{
  __shared__ float sAr[H * D * D];
  __shared__ float sMr[H * D * D];
  for (int i = threadIdx.x; i < H * D * D; i += 256) { sAr[i] = arel[i]; sMr[i] = mrel[i]; }
  __syncthreads();

  int idx = blockIdx.x * 256 + threadIdx.x;
  if (idx >= total) return;
  int hd = idx & (H - 1);
  int node = idx >> 2;

  const float4* k4 = (const float4*)(kin + (size_t)node * C + hd * D);
  const float4* v4 = (const float4*)(vin + (size_t)node * C + hd * D);
  float kk[D], vv[D];
#pragma unroll
  for (int i = 0; i < 4; ++i) {
    float4 f = k4[i]; kk[4*i] = f.x; kk[4*i+1] = f.y; kk[4*i+2] = f.z; kk[4*i+3] = f.w;
    float4 g = v4[i]; vv[4*i] = g.x; vv[4*i+1] = g.y; vv[4*i+2] = g.z; vv[4*i+3] = g.w;
  }
  const float* A = sAr + hd * D * D;
  const float* M = sMr + hd * D * D;
  float* ko = kr + (size_t)node * C + hd * D;
  float* vo = vr + (size_t)node * C + hd * D;
#pragma unroll
  for (int e = 0; e < D; ++e) {
    float sk = 0.0f, sv = 0.0f;
#pragma unroll
    for (int d2 = 0; d2 < D; ++d2) { sk += kk[d2] * A[d2 * D + e]; sv += vv[d2] * M[d2 * D + e]; }
    ko[e] = sk; vo[e] = sv;
  }
}

// ---------------------------------------------------------------------------
// Edge phase 1: alpha[e,h] = (q[dst] . kr[src]) * prel[h] * scale ; atomic max
// ---------------------------------------------------------------------------
__global__ __launch_bounds__(256)
void edge_alpha_kernel(const float* __restrict__ qd, const float* __restrict__ kr,
                       const int* __restrict__ src, const int* __restrict__ dst,
                       const float* __restrict__ prel,
                       float* __restrict__ alpha, unsigned* __restrict__ amax, int total)
{
  int idx = blockIdx.x * 256 + threadIdx.x;
  if (idx >= total) return;
  int hd = idx & (H - 1);
  int e  = idx >> 2;
  int s  = src[e], dd = dst[e];
  const float4* q4 = (const float4*)(qd + (size_t)dd * C + hd * D);
  const float4* k4 = (const float4*)(kr + (size_t)s  * C + hd * D);
  float dot = 0.0f;
#pragma unroll
  for (int i = 0; i < 4; ++i) {
    float4 a = q4[i], b = k4[i];
    dot += a.x * b.x + a.y * b.y + a.z * b.z + a.w * b.w;
  }
  float v = dot * prel[hd] * SCALE;
  alpha[idx] = v;
  atomicMax(&amax[(size_t)dd * H + hd], encf(v));
}

// ---------------------------------------------------------------------------
// Edge phase 2: ev = exp(alpha - max[dst]); alpha <- ev; den[dst] += ev
// ---------------------------------------------------------------------------
__global__ __launch_bounds__(256)
void edge_expden_kernel(float* __restrict__ alpha, const int* __restrict__ dst,
                        const unsigned* __restrict__ amax, float* __restrict__ aden, int total)
{
  int idx = blockIdx.x * 256 + threadIdx.x;
  if (idx >= total) return;
  int hd = idx & (H - 1);
  int e  = idx >> 2;
  int dd = dst[e];
  float m  = decf(amax[(size_t)dd * H + hd]);
  float ev = expf(alpha[idx] - m);
  alpha[idx] = ev;
  atomicAdd(&aden[(size_t)dd * H + hd], ev);
}

// ---------------------------------------------------------------------------
// Edge phase 3: agg[dst] += vr[src] * (ev / (den[dst] + 1e-16))
// ---------------------------------------------------------------------------
__global__ __launch_bounds__(256)
void edge_scatter_kernel(const float* __restrict__ alpha, const int* __restrict__ dst,
                         const int* __restrict__ src, const float* __restrict__ vr,
                         const float* __restrict__ aden, float* __restrict__ agg, int total)
{
  int idx = blockIdx.x * 256 + threadIdx.x;
  if (idx >= total) return;
  int hd = idx & (H - 1);
  int e  = idx >> 2;
  int s  = src[e], dd = dst[e];
  float w = alpha[idx] / (aden[(size_t)dd * H + hd] + 1e-16f);
  const float4* v4 = (const float4*)(vr + (size_t)s * C + hd * D);
  float* out = agg + (size_t)dd * C + hd * D;
#pragma unroll
  for (int i = 0; i < 4; ++i) {
    float4 f = v4[i];
    atomicAdd(&out[4*i + 0], f.x * w);
    atomicAdd(&out[4*i + 1], f.y * w);
    atomicAdd(&out[4*i + 2], f.z * w);
    atomicAdd(&out[4*i + 3], f.w * w);
  }
}

// ---------------------------------------------------------------------------
// Exact GELU (approximate=False), in place
// ---------------------------------------------------------------------------
__global__ __launch_bounds__(256)
void gelu_kernel(float* __restrict__ x, int total)
{
  int idx = blockIdx.x * 256 + threadIdx.x;
  if (idx >= total) return;
  float v = x[idx];
  x[idx] = 0.5f * v * (1.0f + erff(v * 0.70710678118654752f));
}

// ---------------------------------------------------------------------------
// Host-side orchestration (graph-capture safe: only async launches/memsets)
// ---------------------------------------------------------------------------
extern "C" void kernel_launch(void* const* d_in, const int* in_sizes, int n_in,
                              void* d_out, int out_size, void* d_ws, size_t ws_size,
                              hipStream_t stream)
{
  (void)in_sizes; (void)n_in; (void)out_size; (void)ws_size;

  const float* x     = (const float*)d_in[0];
  const int*   ei    = (const int*)  d_in[1];
  const float* lin0W = (const float*)d_in[2];
  const float* lin0b = (const float*)d_in[3];
  const float* kW    = (const float*)d_in[4];
  const float* kb    = (const float*)d_in[5];
  const float* qW    = (const float*)d_in[6];
  const float* qb    = (const float*)d_in[7];
  const float* vW    = (const float*)d_in[8];
  const float* vb    = (const float*)d_in[9];
  const float* aW    = (const float*)d_in[10];
  const float* ab    = (const float*)d_in[11];
  const float* skip  = (const float*)d_in[12];
  const float* arel  = (const float*)d_in[13];
  const float* mrel  = (const float*)d_in[14];
  const float* prel  = (const float*)d_in[15];

  float* h = (float*)d_out;                 // [T,N,C] — final output lives here

  const size_t TNC = (size_t)T * NN * C;
  const size_t NC  = (size_t)NN * C;
  const size_t NH  = (size_t)NN * H;
  const size_t EH  = (size_t)EE * H;

  float* p = (float*)d_ws;
  float* kbuf  = p; p += TNC;
  float* qbuf  = p; p += TNC;
  float* vbuf  = p; p += TNC;
  float* agg   = p; p += TNC;
  float* krb   = p; p += NC;
  float* vrb   = p; p += NC;
  float* alpha = p; p += EH;
  float* aden  = p; p += NH;
  unsigned* amax = (unsigned*)p;            // NH uints

  const dim3 gGemm(NN / 16, T);
  const int edgeBlocks = (int)((EH + 255) / 256);
  const int relBlocks  = (int)((NH + 255) / 256);
  const int geluBlocks = (int)((TNC + 255) / 256);

  // Input projection + ReLU
  gemm64_wmma_kernel<<<gGemm, 128, 0, stream>>>(x, lin0W, lin0b, h,
                                                nullptr, nullptr, NN, 1);

  const int SRC_T[R] = {0, 1};
  const int DST_T[R] = {1, 0};

  for (int l = 0; l < L; ++l) {
    const size_t wOff = (size_t)l * T * C * C;
    const size_t bOff = (size_t)l * T * C;

    gemm64_wmma_kernel<<<gGemm, 128, 0, stream>>>(h, kW + wOff, kb + bOff, kbuf,
                                                  nullptr, nullptr, NN, 0);
    gemm64_wmma_kernel<<<gGemm, 128, 0, stream>>>(h, qW + wOff, qb + bOff, qbuf,
                                                  nullptr, nullptr, NN, 0);
    gemm64_wmma_kernel<<<gGemm, 128, 0, stream>>>(h, vW + wOff, vb + bOff, vbuf,
                                                  nullptr, nullptr, NN, 0);

    hipMemsetAsync(agg, 0, TNC * sizeof(float), stream);

    for (int r = 0; r < R; ++r) {
      const int s = SRC_T[r], d = DST_T[r];
      const size_t relOff = (size_t)(l * R + r) * H * D * D;

      rel_transform_kernel<<<relBlocks, 256, 0, stream>>>(
          kbuf + (size_t)s * NC, vbuf + (size_t)s * NC,
          arel + relOff, mrel + relOff, krb, vrb, (int)NH);

      hipMemsetAsync(amax, 0, NH * sizeof(unsigned), stream);
      hipMemsetAsync(aden, 0, NH * sizeof(float), stream);

      const int* srcI = ei + (size_t)r * 2 * EE;
      const int* dstI = srcI + EE;
      const float* pr = prel + (size_t)(l * R + r) * H;

      edge_alpha_kernel<<<edgeBlocks, 256, 0, stream>>>(
          qbuf + (size_t)d * NC, krb, srcI, dstI, pr, alpha, amax, (int)EH);
      edge_expden_kernel<<<edgeBlocks, 256, 0, stream>>>(
          alpha, dstI, amax, aden, (int)EH);
      edge_scatter_kernel<<<edgeBlocks, 256, 0, stream>>>(
          alpha, dstI, srcI, vrb, aden, agg + (size_t)d * NC, (int)EH);
    }

    gelu_kernel<<<geluBlocks, 256, 0, stream>>>(agg, (int)TNC);

    // out-proj + per-type sigmoid skip blend, fused epilogue (reads+writes h)
    gemm64_wmma_kernel<<<gGemm, 128, 0, stream>>>(agg, aW + wOff, ab + bOff, h,
                                                  h, skip + (size_t)l * T, NN, 2);
  }
}
